// Block_ATC_21689584844856
// MI455X (gfx1250) — matrix operand order repaired and checked by reading, not compiled, not run on hardware
//
#include <hip/hip_runtime.h>
#include <math.h>

// ---------------------------------------------------------------------------
// MI455X (gfx1250, wave32) cosine-distance kernel:
//   metric = x / ||x||_2 (rows of length C), out = 1 - metric @ metric^T
// B=32, T=1024, C=768, all f32.
//
// Strategy: f16 hi/lo split GEMM on v_wmma_f32_16x16x32_f16.
//   m = hi + lo  (hi = f16(m), lo = f16(m - hi))
//   m_i . m_j ~= hi.hi + hi.lo + lo.hi   (lo.lo ~ 2^-24, dropped)
// 3x f16-WMMA work (~155 GFLOP at multi-PFLOP f16 matrix rate) keeps us within
// a small factor of the 9.6us HBM floor while preserving ~f32 accuracy.
// ---------------------------------------------------------------------------

typedef __attribute__((ext_vector_type(16))) _Float16 v16h;
typedef __attribute__((ext_vector_type(8)))  float    v8f;

#define B_  32
#define T_  1024
#define C_  768
#define BM  128
#define BN  128
#define BK  32
#define LDK 40   // padded row stride in halves: 80 bytes (16B aligned, 20 banks)

// -------------------------------- pass 1 -----------------------------------
// One block per row: 1/||x_row||_2 into ws (B*T floats = 128 KB).
__global__ __launch_bounds__(256) void row_rnorm_kernel(
    const float* __restrict__ x, float* __restrict__ rnorm) {
  const int row = blockIdx.x;                 // 0 .. B*T-1
  const float* xr = x + (size_t)row * C_;
  const int t = threadIdx.x;
  float s = 0.0f;
  for (int i = t; i < C_; i += 256) { float v = xr[i]; s += v * v; }
  __shared__ float red[256];
  red[t] = s;
  __syncthreads();
  for (int off = 128; off > 0; off >>= 1) {
    if (t < off) red[t] += red[t + off];
    __syncthreads();
  }
  if (t == 0) rnorm[row] = 1.0f / sqrtf(red[0]);
}

// -------------------------------- pass 2 -----------------------------------
union FragU { v16h v; uint4 q[2]; };   // 32B per-lane WMMA A/B fragment
union H4   { _Float16 h[4]; uint2 u; };

__global__ __launch_bounds__(256) void cosdist_wmma_kernel(
    const float* __restrict__ x, const float* __restrict__ rnorm,
    float* __restrict__ out) {
  // LDS: hi/lo f16 tiles for A (rows m0..m0+127) and B (rows n0..n0+127),
  // K-slab of 32. 4 * 128 * 40 * 2B = 40 KB of the 320 KB WGP LDS.
  __shared__ _Float16 Ahi[BM][LDK], Alo[BM][LDK];
  __shared__ _Float16 Bhi[BN][LDK], Blo[BN][LDK];

  const int b    = blockIdx.z;
  const int m0   = blockIdx.y * BM;
  const int n0   = blockIdx.x * BN;
  const int tid  = threadIdx.x;
  const int lane = tid & 31;
  const int wave = tid >> 5;       // 8 waves
  const int wm   = wave & 3;       // 4 waves along M -> 32 rows each
  const int wn   = wave >> 2;      // 2 waves along N -> 64 cols each
  const int hsel = lane >> 4;      // lane-group (0: lanes 0-15, 1: 16-31)
  const int lidx = lane & 15;

  const float* xb = x + (size_t)b * T_ * C_;
  const float* rn = rnorm + b * T_;

  v8f acc[2][4] = {};               // 2 M-tiles x 4 N-tiles of 16x16 f32

  for (int k0 = 0; k0 < C_; k0 += BK) {
    __syncthreads();                // protect LDS reads of previous k-step

    // ---- stage A and B K-slabs: f32 -> normalize -> f16 hi/lo -> LDS ----
    // 128x32 f32 per matrix = 4096 elems; 256 threads x 4 passes x float4.
    for (int pass = 0; pass < 4; ++pass) {
      const int linear = (pass * 256 + tid) * 4;   // element index in tile
      const int row = linear >> 5;                 // /32
      const int col = linear & 31;

      // Prefetch next k-slab into L2 (global_prefetch_b8).
      if (k0 + BK < C_) {
        __builtin_prefetch(xb + (size_t)(m0 + row) * C_ + (k0 + BK) + col, 0, 1);
        __builtin_prefetch(xb + (size_t)(n0 + row) * C_ + (k0 + BK) + col, 0, 1);
      }

      // A side
      {
        const float4 v = *(const float4*)(xb + (size_t)(m0 + row) * C_ + k0 + col);
        const float r = rn[m0 + row];
        const float m[4] = {v.x * r, v.y * r, v.z * r, v.w * r};
        H4 hh, ll;
#pragma unroll
        for (int e = 0; e < 4; ++e) {
          _Float16 h = (_Float16)m[e];
          hh.h[e] = h;
          ll.h[e] = (_Float16)(m[e] - (float)h);
        }
        *(uint2*)&Ahi[row][col] = hh.u;            // ds_store_b64
        *(uint2*)&Alo[row][col] = ll.u;
      }
      // B side
      {
        const float4 v = *(const float4*)(xb + (size_t)(n0 + row) * C_ + k0 + col);
        const float r = rn[n0 + row];
        const float m[4] = {v.x * r, v.y * r, v.z * r, v.w * r};
        H4 hh, ll;
#pragma unroll
        for (int e = 0; e < 4; ++e) {
          _Float16 h = (_Float16)m[e];
          hh.h[e] = h;
          ll.h[e] = (_Float16)(m[e] - (float)h);
        }
        *(uint2*)&Bhi[row][col] = hh.u;
        *(uint2*)&Blo[row][col] = ll.u;
      }
    }
    __syncthreads();

    // ---- load WMMA fragments from LDS ----
    // A 16x32 f16 (ISA table, interleaved): lane-group 0 holds K{0..7,16..23},
    // group 1 holds K{8..15,24..31}; two ds_read_b128 per fragment.
    FragU ahi[2], alo[2];
#pragma unroll
    for (int i = 0; i < 2; ++i) {
      const int r = wm * 32 + i * 16 + lidx;
      const _Float16* ph = &Ahi[r][hsel * 8];
      const _Float16* pl = &Alo[r][hsel * 8];
      ahi[i].q[0] = *(const uint4*)ph;
      ahi[i].q[1] = *(const uint4*)(ph + 16);
      alo[i].q[0] = *(const uint4*)pl;
      alo[i].q[1] = *(const uint4*)(pl + 16);
    }
    // B 32x16 f16 (same interleave as A): lane-group 0 holds K{0..7,16..23},
    // group 1 holds K{8..15,24..31}; lane index = N column = row n.
    FragU bhi[4], blo[4];
#pragma unroll
    for (int j = 0; j < 4; ++j) {
      const int r = wn * 64 + j * 16 + lidx;
      const _Float16* ph = &Bhi[r][hsel * 8];
      const _Float16* pl = &Blo[r][hsel * 8];
      bhi[j].q[0] = *(const uint4*)ph;
      bhi[j].q[1] = *(const uint4*)(ph + 16);
      blo[j].q[0] = *(const uint4*)pl;
      blo[j].q[1] = *(const uint4*)(pl + 16);
    }

    // ---- 2x4 tiles x 3 split-terms = 24 WMMAs per wave per k-step ----
#pragma unroll
    for (int i = 0; i < 2; ++i) {
#pragma unroll
      for (int j = 0; j < 4; ++j) {
        acc[i][j] = __builtin_amdgcn_wmma_f32_16x16x32_f16(
            false, ahi[i].v, false, bhi[j].v, (short)0, acc[i][j], false, false);
        acc[i][j] = __builtin_amdgcn_wmma_f32_16x16x32_f16(
            false, ahi[i].v, false, blo[j].v, (short)0, acc[i][j], false, false);
        acc[i][j] = __builtin_amdgcn_wmma_f32_16x16x32_f16(
            false, alo[i].v, false, bhi[j].v, (short)0, acc[i][j], false, false);
      }
    }
  }

  // ---- epilogue: out = 1 - acc ----
  // C/D 16x16 f32 layout: lanes 0-15 VGPR v -> (M=v, N=lane);
  // lanes 16-31 VGPR v -> (M=8+v, N=lane-16).
  float* ob = out + ((size_t)b * T_ + m0) * T_ + n0;
#pragma unroll
  for (int i = 0; i < 2; ++i) {
#pragma unroll
    for (int j = 0; j < 4; ++j) {
      const int rbase = wm * 32 + i * 16 + hsel * 8;
      const int cc    = wn * 64 + j * 16 + lidx;
#pragma unroll
      for (int v = 0; v < 8; ++v) {
        ob[(size_t)(rbase + v) * T_ + cc] = 1.0f - acc[i][j][v];
      }
    }
  }
}

// ------------------------------ launcher -----------------------------------
extern "C" void kernel_launch(void* const* d_in, const int* in_sizes, int n_in,
                              void* d_out, int out_size, void* d_ws, size_t ws_size,
                              hipStream_t stream) {
  (void)in_sizes; (void)n_in; (void)out_size; (void)ws_size;
  const float* x   = (const float*)d_in[0];          // [B,T,C] f32
  float* out       = (float*)d_out;                  // [B,T,T] f32
  float* rnorm     = (float*)d_ws;                   // B*T floats (128 KB)

  row_rnorm_kernel<<<dim3(B_ * T_), 256, 0, stream>>>(x, rnorm);

  dim3 grid(T_ / BN, T_ / BM, B_);                   // 8 x 8 x 32 blocks
  cosdist_wmma_kernel<<<grid, 256, 0, stream>>>(x, rnorm, out);
}
